// PacketSwitchingPINN_2104533975560
// MI455X (gfx1250) — matrix operand order repaired and checked by reading, not compiled, not run on hardware
//
#include <hip/hip_runtime.h>

typedef unsigned short u16;
typedef __attribute__((ext_vector_type(16))) __bf16 v16bf;
typedef __attribute__((ext_vector_type(8)))  __bf16 v8bf;
typedef __attribute__((ext_vector_type(8)))  float  v8f;

#define B_TOK   4096
#define IN_DIM  4096
#define NBLK    8
#define BLK     512
#define NEXP    6
#define H1N     1024
#define H2N     512

#define H1_STRIDE  1032   // 1024 + 8 bf16 pad -> 4-bank rotation per row
#define STG_STRIDE 40     // 32 + 8 bf16 pad

__device__ __forceinline__ u16 f2bf(float f) {
  union { float f; unsigned u; } v; v.f = f;
  unsigned u = v.u;
  u += 0x7fffu + ((u >> 16) & 1u);       // round-to-nearest-even
  return (u16)(u >> 16);
}

__device__ __forceinline__ float fast_tanh(float x) {
  float e = __expf(2.0f * x);
  return (e - 1.0f) / (e + 1.0f);
}

__device__ __forceinline__ v8f zero8() {
  v8f z;
#pragma unroll
  for (int i = 0; i < 8; ++i) z[i] = 0.0f;
  return z;
}

__device__ __forceinline__ v16bf combine16(v8bf lo, v8bf hi) {
  v16bf r;
#pragma unroll
  for (int i = 0; i < 8; ++i) { r[i] = lo[i]; r[i + 8] = hi[i]; }
  return r;
}

// Build a 16-element bf16 fragment from two contiguous 16B chunks (loads at use).
__device__ __forceinline__ v16bf frag16(const u16* p0, const u16* p1) {
  return combine16(*(const v8bf*)p0, *(const v8bf*)p1);
}

// ---- CDNA5 async global->LDS copy (ASYNCcnt-tracked), ISA 08_async_tensor §4 ----
__device__ __forceinline__ void async_b128(unsigned lds_off, const u16* gptr) {
  asm volatile("global_load_async_to_lds_b128 %0, %1, off"
               :: "v"(lds_off), "v"((unsigned long long)(uintptr_t)gptr)
               : "memory");
}

__device__ __forceinline__ void wait_async0() {
#if __has_builtin(__builtin_amdgcn_s_wait_asynccnt)
  __builtin_amdgcn_s_wait_asynccnt(0);
#else
  asm volatile("s_wait_asynccnt 0x0" ::: "memory");
#endif
}

// ---------------- precision / layout conversion ----------------

__global__ void k_cvt_x(const float* __restrict__ x, u16* __restrict__ xb, int n) {
  int i = blockIdx.x * blockDim.x + threadIdx.x;
  int stride = gridDim.x * blockDim.x;
  for (; i < n; i += stride) xb[i] = f2bf(x[i]);
}

// We1 [E][512 k][1024 n] f32 -> We1T [E][1024 n][512 k] bf16
__global__ void k_t_we1(const float* __restrict__ w, u16* __restrict__ wt) {
  int i = blockIdx.x * blockDim.x + threadIdx.x;
  if (i >= NEXP * H1N * BLK) return;
  int e = i / (H1N * BLK);
  int r = i % (H1N * BLK);
  int n = r / BLK;
  int k = r % BLK;
  wt[i] = f2bf(w[((size_t)e * BLK + k) * H1N + n]);
}

// We2 [E][1024 k][512 n] f32 -> We2T [E][512 n][1024 k] bf16
__global__ void k_t_we2(const float* __restrict__ w, u16* __restrict__ wt) {
  int i = blockIdx.x * blockDim.x + threadIdx.x;
  if (i >= NEXP * H2N * H1N) return;
  int e = i / (H2N * H1N);
  int r = i % (H2N * H1N);
  int n = r / H1N;
  int k = r % H1N;
  wt[i] = f2bf(w[((size_t)e * H1N + k) * H2N + n]);
}

// ---------------- router: one wave per (b, s) ----------------

__global__ void k_router(const float* __restrict__ x,
                         const float* __restrict__ Wr1, const float* __restrict__ br1,
                         const float* __restrict__ Wr2, const float* __restrict__ br2,
                         float* __restrict__ routing) {
  int wave = threadIdx.x >> 5;
  int lane = threadIdx.x & 31;
  int pair = blockIdx.x * 8 + wave;          // 4096 blocks * 8 waves = 32768 pairs
  int b = pair >> 3, s = pair & 7;

  const float* xr = x + (size_t)b * IN_DIM + s * BLK;
  const float* w1 = Wr1 + (size_t)s * BLK * 32 + lane;
  float acc = br1[s * 32 + lane];
  for (int d = 0; d < BLK; ++d) acc = fmaf(xr[d], w1[(size_t)d * 32], acc);
  float r1 = fmaxf(acc, 0.0f);

  float lg[6];
#pragma unroll
  for (int e = 0; e < 6; ++e) {
    float p = r1 * Wr2[(s * 32 + lane) * 6 + e];
#pragma unroll
    for (int off = 16; off >= 1; off >>= 1) p += __shfl_xor(p, off, 32);
    lg[e] = br2[s * 6 + e] + p;
  }
  float m = lg[0];
#pragma unroll
  for (int e = 1; e < 6; ++e) m = fmaxf(m, lg[e]);
  float ex[6]; float den = 0.0f;
#pragma unroll
  for (int e = 0; e < 6; ++e) { ex[e] = __expf(lg[e] - m); den += ex[e]; }
  if (lane == 0) {
    float inv = 1.0f / den;
    float* o = routing + ((size_t)b * 8 + s) * 6;
#pragma unroll
    for (int e = 0; e < 6; ++e) o[e] = ex[e] * inv;
  }
}

// ---------------- fused expert stack (WMMA + async double-buffered staging) ----
// Grid: (64 row-tiles of 64 tokens, 48 = s*6+e). Block: 512 threads = 16 waves.
// Each wave owns ALL 64 rows (4 row-groups) and a 1/16 slice of N, so every
// B fragment loaded from LDS feeds 4 independent WMMAs (one per row-group).

// Stage We1T[:, k1*32 .. +32) -> slab (1024 rows x 32 bf16). 8 async b128 / wave.
__device__ __forceinline__ void stage1_async(const u16* w1base, unsigned stg_off,
                                             int k1, int wave, int lane) {
  int rsel = lane >> 2, csel = lane & 3;
  int row = wave * 64 + rsel;
  const u16* g = w1base + (size_t)row * BLK + k1 * 32 + csel * 8;
  unsigned l = stg_off + (unsigned)(row * STG_STRIDE + csel * 8) * 2;
#pragma unroll 1
  for (int j = 0; j < 8; ++j) {
    async_b128(l, g);
    g += 8 * BLK;                 // +8 rows in We1T
    l += 8 * STG_STRIDE * 2;      // +8 rows in slab (bytes)
  }
}

// Stage We2T[:, k2*32 .. +32) -> slab (512 rows x 32 bf16). 4 async b128 / wave.
__device__ __forceinline__ void stage2_async(const u16* w2base, unsigned stg_off,
                                             int k2, int wave, int lane) {
  int rsel = lane >> 2, csel = lane & 3;
  int row = wave * 32 + rsel;
  const u16* g = w2base + (size_t)row * H1N + k2 * 32 + csel * 8;
  unsigned l = stg_off + (unsigned)(row * STG_STRIDE + csel * 8) * 2;
#pragma unroll 1
  for (int j = 0; j < 4; ++j) {
    async_b128(l, g);
    g += 8 * H1N;
    l += 8 * STG_STRIDE * 2;
  }
}

__global__ __launch_bounds__(512)
void k_expert(const u16* __restrict__ xb, const u16* __restrict__ we1t,
              const float* __restrict__ be1, const u16* __restrict__ we2t,
              const float* __restrict__ be2, const float* __restrict__ we3,
              const float* __restrict__ be3, const float* __restrict__ routing,
              float* __restrict__ eo_ws) {
  extern __shared__ char smem[];
  u16* h1   = (u16*)smem;                       // [64][H1_STRIDE] bf16   (129 KB)
  u16* stg0 = h1 + 64 * H1_STRIDE;              // slab buf 0 [1024][40]  (80 KB)
  u16* stg1 = stg0 + 1024 * STG_STRIDE;         // slab buf 1 [1024][40]  (80 KB)
  float* eo_p = (float*)stg0;                   // reused at the end: [16][64][3]

  const unsigned stg_off0 = (unsigned)(uintptr_t)stg0;
  const unsigned stg_off1 = (unsigned)(uintptr_t)stg1;

  const int tid  = threadIdx.x;
  const int wave = tid >> 5;                   // 0..15: N-slice owner
  const int lane = tid & 31;
  const int hl   = lane >> 4;                  // half-wave select
  const int lr   = lane & 15;
  const int s    = blockIdx.y / 6;
  const int e    = blockIdx.y % 6;
  const int row0 = blockIdx.x * 64;

  // ---------- GEMM1: h1 = tanh(X_s @ We1 + be1)  (64 x 1024) ----------
  v8f acc1[16];                                // [r*4 + t]: row-group r, n-tile t
#pragma unroll
  for (int t = 0; t < 16; ++t) acc1[t] = zero8();

  const size_t aRow = (size_t)(row0 + lr) * IN_DIM + s * BLK + hl * 8;
  const u16* w1base = we1t + (size_t)e * H1N * BLK;

  stage1_async(w1base, stg_off0, 0, wave, lane);       // prologue: slab k1=0
  v8bf alo[4], ahi[4];                                 // prefetch A(k1=0), 4 row-grps
#pragma unroll
  for (int r = 0; r < 4; ++r) {
    const u16* p = xb + aRow + (size_t)r * 16 * IN_DIM;
    alo[r] = *(const v8bf*)p; ahi[r] = *(const v8bf*)(p + 16);
  }

  for (int k1 = 0; k1 < 16; ++k1) {
    wait_async0();                 // own slab batch landed in LDS
    __syncthreads();               // everyone's batch landed; other buf free
    if (k1 + 1 < 16) stage1_async(w1base, ((k1 + 1) & 1) ? stg_off1 : stg_off0,
                                  k1 + 1, wave, lane);
    v16bf a[4];
#pragma unroll
    for (int r = 0; r < 4; ++r) a[r] = combine16(alo[r], ahi[r]);
    if (k1 + 1 < 16) {             // prefetch next A fragments (overlap WMMAs)
#pragma unroll
      for (int r = 0; r < 4; ++r) {
        const u16* p = xb + aRow + (size_t)r * 16 * IN_DIM + (k1 + 1) * 32;
        alo[r] = *(const v8bf*)p; ahi[r] = *(const v8bf*)(p + 16);
      }
    }
    const u16* sb  = (k1 & 1) ? stg1 : stg0;
    const u16* bp0 = sb + (wave * 64 + lr) * STG_STRIDE + hl * 8;
#pragma unroll
    for (int t = 0; t < 4; ++t) {              // one B fragment -> 4 WMMAs
      const u16* q = bp0 + t * (16 * STG_STRIDE);
      v16bf bb = frag16(q, q + 16);
#pragma unroll
      for (int r = 0; r < 4; ++r)
        acc1[r * 4 + t] = __builtin_amdgcn_wmma_f32_16x16x32_bf16(
            false, a[r], false, bb, (short)0, acc1[r * 4 + t], false, false);
    }
  }

  // activation, write h1 tile to LDS (K-contiguous for GEMM2 A-fragments)
#pragma unroll
  for (int t = 0; t < 4; ++t) {
    int n = (wave * 4 + t) * 16 + lr;
    float bias = be1[e * H1N + n];
#pragma unroll
    for (int r = 0; r < 4; ++r) {
#pragma unroll
      for (int i = 0; i < 8; ++i) {
        int lrow = r * 16 + hl * 8 + i;
        h1[lrow * H1_STRIDE + n] = f2bf(fast_tanh(acc1[r * 4 + t][i] + bias));
      }
    }
  }
  __syncthreads();                 // h1 complete everywhere; slab bufs free

  // ---------- GEMM2: h2 = tanh(h1 @ We2 + be2), fused We3 projection ----------
  v8f acc2[8];                                 // [r*2 + t]
#pragma unroll
  for (int t = 0; t < 8; ++t) acc2[t] = zero8();

  const u16* w2base = we2t + (size_t)e * H2N * H1N;
  const u16* h1row  = h1 + lr * H1_STRIDE + hl * 8;   // + r*16*H1_STRIDE + k2*32

  stage2_async(w2base, stg_off0, 0, wave, lane);       // prologue: slab k2=0

  for (int k2 = 0; k2 < 32; ++k2) {
    wait_async0();
    __syncthreads();
    if (k2 + 1 < 32) stage2_async(w2base, ((k2 + 1) & 1) ? stg_off1 : stg_off0,
                                  k2 + 1, wave, lane);
    v16bf a[4];
#pragma unroll
    for (int r = 0; r < 4; ++r) {
      const u16* p = h1row + (r * 16) * H1_STRIDE + k2 * 32;
      a[r] = frag16(p, p + 16);
    }
    const u16* sb  = (k2 & 1) ? stg1 : stg0;
    const u16* bp0 = sb + (wave * 32 + lr) * STG_STRIDE + hl * 8;
#pragma unroll
    for (int t = 0; t < 2; ++t) {              // one B fragment -> 4 WMMAs
      const u16* q = bp0 + t * (16 * STG_STRIDE);
      v16bf bb = frag16(q, q + 16);
#pragma unroll
      for (int r = 0; r < 4; ++r)
        acc2[r * 2 + t] = __builtin_amdgcn_wmma_f32_16x16x32_bf16(
            false, a[r], false, bb, (short)0, acc2[r * 2 + t], false, false);
    }
  }

  __syncthreads();                      // all slab reads done; reuse stg0 as eo_p

  // tanh + per-lane partial eo = h2 @ We3, processed one row-group at a time
#pragma unroll
  for (int r = 0; r < 4; ++r) {
    float ap0[8], ap1[8], ap2[8];
#pragma unroll
    for (int i = 0; i < 8; ++i) { ap0[i] = 0.f; ap1[i] = 0.f; ap2[i] = 0.f; }
#pragma unroll
    for (int t = 0; t < 2; ++t) {
      int n = (wave * 2 + t) * 16 + lr;
      float bias = be2[e * H2N + n];
      const float* w3 = we3 + ((size_t)e * H2N + n) * 3;
      float w30 = w3[0], w31 = w3[1], w32 = w3[2];
#pragma unroll
      for (int i = 0; i < 8; ++i) {
        float h = fast_tanh(acc2[r * 2 + t][i] + bias);
        ap0[i] = fmaf(h, w30, ap0[i]);
        ap1[i] = fmaf(h, w31, ap1[i]);
        ap2[i] = fmaf(h, w32, ap2[i]);
      }
    }
    // reduce across the 16 lanes of each half-wave (K over columns)
#pragma unroll
    for (int off = 8; off >= 1; off >>= 1) {
#pragma unroll
      for (int i = 0; i < 8; ++i) {
        ap0[i] += __shfl_xor(ap0[i], off, 16);
        ap1[i] += __shfl_xor(ap1[i], off, 16);
        ap2[i] += __shfl_xor(ap2[i], off, 16);
      }
    }
    if (lr == 0) {                      // lanes 0 and 16 hold the half-wave sums
#pragma unroll
      for (int i = 0; i < 8; ++i) {
        int lrow = r * 16 + hl * 8 + i;
        float* dst = eo_p + (wave * 64 + lrow) * 3;
        dst[0] = ap0[i]; dst[1] = ap1[i]; dst[2] = ap2[i];
      }
    }
  }
  __syncthreads();
  if (tid < 192) {                      // 64 rows x 3 outputs, deterministic sum
    int lrow = tid / 3, o = tid % 3;
    float v = 0.f;
#pragma unroll
    for (int w = 0; w < 16; ++w) v += eo_p[(w * 64 + lrow) * 3 + o];
    int bg = row0 + lrow;
    float rte = routing[((size_t)bg * 8 + s) * 6 + e];
    eo_ws[(((size_t)(s * 6 + e)) * B_TOK + bg) * 3 + o] = rte * (v + be3[e * 3 + o]);
  }
}

// ---------------- final: mean over blocks + tiny MLP head ----------------

__global__ void k_final(const float* __restrict__ eo_ws,
                        const float* __restrict__ Wa1, const float* __restrict__ ba1,
                        const float* __restrict__ Wa2, const float* __restrict__ ba2,
                        float* __restrict__ out) {
  int b = blockIdx.x * blockDim.x + threadIdx.x;
  if (b >= B_TOK) return;
  float f0 = 0.f, f1 = 0.f, f2 = 0.f;
  for (int p = 0; p < NBLK * NEXP; ++p) {
    const float* src = eo_ws + ((size_t)p * B_TOK + b) * 3;
    f0 += src[0]; f1 += src[1]; f2 += src[2];
  }
  f0 *= 0.125f; f1 *= 0.125f; f2 *= 0.125f;   // mean over 8 blocks
  float o0 = ba2[0], o1 = ba2[1], o2 = ba2[2];
#pragma unroll
  for (int j = 0; j < 16; ++j) {
    float a = fmaxf(ba1[j] + f0 * Wa1[j] + f1 * Wa1[16 + j] + f2 * Wa1[32 + j], 0.0f);
    o0 = fmaf(a, Wa2[j * 3 + 0], o0);
    o1 = fmaf(a, Wa2[j * 3 + 1], o1);
    o2 = fmaf(a, Wa2[j * 3 + 2], o2);
  }
  float* dst = out + (size_t)b * 3;
  dst[0] = o0; dst[1] = o1; dst[2] = o2;
}

// ---------------- launch ----------------

extern "C" void kernel_launch(void* const* d_in, const int* in_sizes, int n_in,
                              void* d_out, int out_size, void* d_ws, size_t ws_size,
                              hipStream_t stream) {
  const float* x   = (const float*)d_in[0];
  const float* Wr1 = (const float*)d_in[1];
  const float* br1 = (const float*)d_in[2];
  const float* Wr2 = (const float*)d_in[3];
  const float* br2 = (const float*)d_in[4];
  const float* We1 = (const float*)d_in[5];
  const float* be1 = (const float*)d_in[6];
  const float* We2 = (const float*)d_in[7];
  const float* be2 = (const float*)d_in[8];
  const float* We3 = (const float*)d_in[9];
  const float* be3 = (const float*)d_in[10];
  const float* Wa1 = (const float*)d_in[11];
  const float* ba1 = (const float*)d_in[12];
  const float* Wa2 = (const float*)d_in[13];
  const float* ba2 = (const float*)d_in[14];
  float* out = (float*)d_out;

  char* ws = (char*)d_ws;
  u16* xb       = (u16*)ws;  ws += (size_t)B_TOK * IN_DIM * 2;           // 32 MB
  u16* we1t     = (u16*)ws;  ws += (size_t)NEXP * H1N * BLK * 2;         // 6 MB
  u16* we2t     = (u16*)ws;  ws += (size_t)NEXP * H2N * H1N * 2;         // 6 MB
  float* routing = (float*)ws; ws += (size_t)B_TOK * NBLK * NEXP * 4;    // 0.75 MB
  float* eo_ws   = (float*)ws;                                           // 2.25 MB

  k_cvt_x<<<4096, 256, 0, stream>>>(x, xb, B_TOK * IN_DIM);
  int t1 = NEXP * H1N * BLK;
  k_t_we1<<<(t1 + 255) / 256, 256, 0, stream>>>(We1, we1t);
  int t2 = NEXP * H2N * H1N;
  k_t_we2<<<(t2 + 255) / 256, 256, 0, stream>>>(We2, we2t);
  k_router<<<4096, 256, 0, stream>>>(x, Wr1, br1, Wr2, br2, routing);

  size_t smem = (size_t)(64 * H1_STRIDE + 2 * 1024 * STG_STRIDE) * 2;    // 295,936 B
  k_expert<<<dim3(64, 48), 512, smem, stream>>>(xb, we1t, be1, we2t, be2,
                                                We3, be3, routing, eo_ws);
  k_final<<<16, 256, 0, stream>>>(eo_ws, Wa1, ba1, Wa2, ba2, out);
}
